// Block_36807869727037
// MI455X (gfx1250) — matrix operand order repaired and checked by reading, not compiled, not run on hardware
//
#include <hip/hip_runtime.h>
#include <hip/hip_bf16.h>

// ---------------------------------------------------------------------------
// GPT block forward on MI455X (gfx1250, wave32, WMMA f16 -> f32 accum)
// B=4, T=2048, C=1024, H=16, d=64
// GEMMs: double-buffered global_load_async_to_lds_b128 pipeline + WMMA,
//        128x256 block tile, 64x64 wave tile (16 WMMA per K-step per wave).
// ---------------------------------------------------------------------------

typedef __attribute__((ext_vector_type(16))) _Float16 v16h;
typedef __attribute__((ext_vector_type(8)))  _Float16 v8h;
typedef __attribute__((ext_vector_type(8)))  float    v8f;

#define DEVINL __device__ __forceinline__

constexpr int   Bdim   = 4;
constexpr int   Tdim   = 2048;
constexpr int   Cdim   = 1024;
constexpr int   Hn     = 16;
constexpr int   Dh     = 64;
constexpr int   ThreeC = 3 * Cdim;
constexpr int   FourC  = 4 * Cdim;
constexpr long  Mrows  = (long)Bdim * Tdim;      // 8192
constexpr int   LDST   = 40;                     // padded LDS row stride (halves)

// ---------------------------------------------------------------------------
// Fragment loaders following CDNA5 ISA 7.12.2 (wave32, 16-bit operands).
// A 16x32 (MxK): lane L -> m=L&15, half=L>>4; elems 0..7 = K[half*8 .. +7],
//                elems 8..15 = K[16+half*8 .. +7]
// B 32x16 (KxN): lane L -> n=L&15, half=L>>4; elems 0..7 = K[half*8 .. +7],
//                elems 8..15 = K[16+half*8 .. +7]
// C/D 16x16:     lane L -> n=L&15; VGPR r -> m = r + 8*(L>>4)
// ---------------------------------------------------------------------------
DEVINL v16h load_frag_a(const _Float16* base /* &A[m0][k0] */, long ld, int lane) {
    const int m = lane & 15, half = lane >> 4;
    const _Float16* p = base + (long)m * ld + half * 8;
    v8h lo = *(const v8h*)(p);
    v8h hi = *(const v8h*)(p + 16);
    v16h r;
#pragma unroll
    for (int i = 0; i < 8; ++i) { r[i] = lo[i]; r[i + 8] = hi[i]; }
    return r;
}

DEVINL v16h load_frag_b(const _Float16* base /* &W[n0][k0] */, long ld, int lane) {
    const int n = lane & 15, half = lane >> 4;
    const _Float16* p = base + (long)n * ld + half * 8;
    v8h lo = *(const v8h*)(p);
    v8h hi = *(const v8h*)(p + 16);
    v16h r;
#pragma unroll
    for (int i = 0; i < 8; ++i) { r[i] = lo[i]; r[i + 8] = hi[i]; }
    return r;
}

DEVINL v8f wmma_f16(v16h a, v16h b, v8f c) {
    return __builtin_amdgcn_wmma_f32_16x16x32_f16(false, a, false, b, (short)0, c,
                                                  false, false);
}

DEVINL float gelu_tanh(float x) {
    const float c = 0.7978845608028654f;
    return 0.5f * x * (1.0f + tanhf(c * (x + 0.044715f * x * x * x)));
}

// Async DMA: 16 bytes global -> LDS, tracked by ASYNCcnt.
// VDST VGPR carries the LDS byte address (low 32 bits of generic shared ptr).
DEVINL void async_copy_b128(unsigned lds_addr, const void* gptr) {
    unsigned long long ga = (unsigned long long)(uintptr_t)gptr;
    asm volatile("global_load_async_to_lds_b128 %0, %1, off"
                 :: "v"(lds_addr), "v"(ga) : "memory");
}
DEVINL unsigned lds_lo32(const void* p) { return (unsigned)(uintptr_t)p; }

// ---------------------------------------------------------------------------
// Elementwise helpers
// ---------------------------------------------------------------------------
__global__ void cast_f32_to_f16_kernel(const float* __restrict__ in,
                                       _Float16* __restrict__ out, long n) {
    long i = (long)blockIdx.x * blockDim.x + threadIdx.x;
    if (i < n) out[i] = (_Float16)in[i];
}

// V transpose: vt[b][h][j][t] = qkv[(b*T + t)*3C + 2C + h*Dh + j]
__global__ void vtrans_kernel(const _Float16* __restrict__ qkv,
                              _Float16* __restrict__ vt) {
    long idx = (long)blockIdx.x * blockDim.x + threadIdx.x;  // exact launch
    int  t   = (int)(idx % Tdim);
    long r   = idx / Tdim;
    int  j   = (int)(r % Dh);   r /= Dh;
    int  h   = (int)(r % Hn);
    int  b   = (int)(r / Hn);
    vt[idx] = qkv[((long)(b * Tdim + t)) * ThreeC + 2 * Cdim + h * Dh + j];
}

// LayerNorm (no gamma/beta, eps=1e-10, biased var), f32 in -> f16 out.
__global__ __launch_bounds__(256)
void ln_f16_kernel(const float* __restrict__ x, _Float16* __restrict__ o) {
    __shared__ float r1[256];
    __shared__ float r2[256];
    const long row = blockIdx.x;
    const float* xr = x + row * Cdim;
    const int tid = threadIdx.x;
    float v[4], s = 0.f, s2 = 0.f;
#pragma unroll
    for (int i = 0; i < 4; ++i) {
        v[i] = xr[tid + i * 256];
        s  += v[i];
        s2 += v[i] * v[i];
    }
    r1[tid] = s; r2[tid] = s2;
    __syncthreads();
    for (int off = 128; off > 0; off >>= 1) {
        if (tid < off) { r1[tid] += r1[tid + off]; r2[tid] += r2[tid + off]; }
        __syncthreads();
    }
    const float mean = r1[0] * (1.0f / Cdim);
    const float msq  = r2[0] * (1.0f / Cdim);
    const float inv  = rsqrtf(msq - mean * mean + 1e-10f);
    _Float16* orow = o + row * Cdim;
#pragma unroll
    for (int i = 0; i < 4; ++i)
        orow[tid + i * 256] = (_Float16)((v[i] - mean) * inv);
}

// ---------------------------------------------------------------------------
// WMMA GEMM: out[M,N] = A[M,K] * W[N,K]^T + bias (+residual / gelu)
// 256 threads = 8 waves; block tile 128x256, wave tile 64x64 (2x4 wave grid).
// Double-buffered async-to-LDS staging (A: 128x32, B: 256x32 per K-step).
// MODE 0: f16 out = acc+bias; MODE 1: f16 out = gelu(acc+bias);
// MODE 2: f32 out = acc+bias+resid
// ---------------------------------------------------------------------------
template <int MODE>
__global__ __launch_bounds__(256)
void wmma_gemm_kernel(const _Float16* __restrict__ A, int lda,
                      const _Float16* __restrict__ W, int ldw,
                      const float* __restrict__ bias,
                      const float* __restrict__ resid,
                      void* __restrict__ outp, int ldo, int K) {
    __shared__ _Float16 ldsA[2][128][LDST];   // 20 KB
    __shared__ _Float16 ldsB[2][256][LDST];   // 40 KB

    const int tid  = threadIdx.x;
    const int lane = tid & 31;
    const int wave = tid >> 5;
    const int wm = wave >> 2, wn = wave & 3;          // 2 x 4 wave grid
    const long m0b = (long)blockIdx.y * 128;
    const long n0b = (long)blockIdx.x * 256;
    const long m0 = m0b + wm * 64;
    const long n0 = n0b + wn * 64;

    // Cooperative fill: A rows 0..127 (2 thr/row, 2 chunks each),
    //                   B rows 0..255 (1 thr/row, 4 chunks each). 6 ops/thread.
    const int ar = tid >> 1;            // A tile row
    const int acoff = (tid & 1) * 16;   // A half-column offset 0 / 16

    auto issue_fill = [&](int buf, int k0) {
        const _Float16* ga = A + (m0b + ar) * lda + k0 + acoff;
        const unsigned la = lds_lo32(&ldsA[buf][ar][acoff]);
        async_copy_b128(la,      ga);
        async_copy_b128(la + 16, ga + 8);
        const _Float16* gb = W + (n0b + tid) * ldw + k0;
        const unsigned lb = lds_lo32(&ldsB[buf][tid][0]);
        async_copy_b128(lb,      gb);
        async_copy_b128(lb + 16, gb + 8);
        async_copy_b128(lb + 32, gb + 16);
        async_copy_b128(lb + 48, gb + 24);
    };

    v8f acc[4][4];
#pragma unroll
    for (int i = 0; i < 4; ++i)
#pragma unroll
        for (int j = 0; j < 4; ++j) acc[i][j] = {};

    const int nk = K >> 5;
    issue_fill(0, 0);
    for (int it = 0; it < nk; ++it) {
        const int cur = it & 1;
        if (it + 1 < nk) {
            issue_fill(cur ^ 1, (it + 1) << 5);
            // newest 6 async ops (next buffer) may remain in flight
            asm volatile("s_wait_asynccnt 0x6" ::: "memory");
        } else {
            asm volatile("s_wait_asynccnt 0x0" ::: "memory");
        }
        __syncthreads();   // all waves' chunks of buffer `cur` are in LDS

        v16h a[4];
#pragma unroll
        for (int i = 0; i < 4; ++i)
            a[i] = load_frag_a(&ldsA[cur][wm * 64 + i * 16][0], LDST, lane);
#pragma unroll
        for (int j = 0; j < 4; ++j) {
            v16h b = load_frag_b(&ldsB[cur][wn * 64 + j * 16][0], LDST, lane);
#pragma unroll
            for (int i = 0; i < 4; ++i)
                acc[i][j] = wmma_f16(a[i], b, acc[i][j]);
        }
        __syncthreads();   // done reading `cur` before it is refilled
    }

    const int n = lane & 15, half = lane >> 4;
#pragma unroll
    for (int i = 0; i < 4; ++i) {
#pragma unroll
        for (int j = 0; j < 4; ++j) {
            const long col = n0 + j * 16 + n;
            const float bv = bias[col];
#pragma unroll
            for (int r = 0; r < 8; ++r) {
                const long row = m0 + i * 16 + r + 8 * half;
                float val = acc[i][j][r] + bv;
                if constexpr (MODE == 0) {
                    ((_Float16*)outp)[row * ldo + col] = (_Float16)val;
                } else if constexpr (MODE == 1) {
                    ((_Float16*)outp)[row * ldo + col] = (_Float16)gelu_tanh(val);
                } else {
                    ((float*)outp)[row * ldo + col] = val + resid[row * ldo + col];
                }
            }
        }
    }
}

// ---------------------------------------------------------------------------
// Causal flash attention. One wave per 16 queries; 8 waves/block -> 128 q.
// qkv: f16 [B*T, 3C] (q at 0, k at +C, v at +2C). vt: f16 [B,H,Dh,T].
// y: f16 [B*T, C].
// ---------------------------------------------------------------------------
__global__ __launch_bounds__(256)
void attn_kernel(const _Float16* __restrict__ qkv,
                 const _Float16* __restrict__ vt,
                 _Float16* __restrict__ y) {
    __shared__ _Float16 plds[8 * 16 * 32];   // per-wave 16x32 P staging
    const int lane = threadIdx.x & 31;
    const int wave = threadIdx.x >> 5;
    const int b = blockIdx.z, h = blockIdx.y;
    const int q0 = blockIdx.x * 128 + wave * 16;
    const int n = lane & 15, half = lane >> 4;

    const _Float16* qbase = qkv + ((long)(b * Tdim + q0)) * ThreeC + h * Dh;
    v16h aq0 = load_frag_a(qbase,      ThreeC, lane);   // K-dim 0..31
    v16h aq1 = load_frag_a(qbase + 32, ThreeC, lane);   // K-dim 32..63

    v8f o[4];
#pragma unroll
    for (int j = 0; j < 4; ++j) o[j] = {};
    float mrun[8], lrun[8];
#pragma unroll
    for (int r = 0; r < 8; ++r) { mrun[r] = -3.0e38f; lrun[r] = 0.f; }

    _Float16* myp = plds + wave * (16 * 32);
    const int qmax = q0 + 15;

    for (int kk = 0; kk <= qmax; kk += 32) {
        // ---- S = Q * K^T for 16 x 32 tile (two 16x16 halves) -------------
        v8f s[2];
#pragma unroll
        for (int t = 0; t < 2; ++t) {
            const _Float16* kb =
                qkv + ((long)(b * Tdim + kk + t * 16)) * ThreeC + Cdim + h * Dh;
            v16h bk0 = load_frag_b(kb,      ThreeC, lane);
            v16h bk1 = load_frag_b(kb + 32, ThreeC, lane);
            v8f z = {};
            s[t] = wmma_f16(aq0, bk0, z);
            s[t] = wmma_f16(aq1, bk1, s[t]);
        }
        // ---- scale + causal mask ----------------------------------------
#pragma unroll
        for (int t = 0; t < 2; ++t) {
#pragma unroll
            for (int r = 0; r < 8; ++r) {
                const int key = kk + t * 16 + n;
                const int qr  = q0 + r + 8 * half;
                float v = s[t][r] * 0.125f;          // 1/sqrt(64)
                if (key > qr) v = -3.0e38f;
                s[t][r] = v;
            }
        }
        // ---- online softmax stats (row r lives in reg r across 16 lanes) -
#pragma unroll
        for (int r = 0; r < 8; ++r) {
            float cm = fmaxf(s[0][r], s[1][r]);
#pragma unroll
            for (int msk = 1; msk < 16; msk <<= 1)
                cm = fmaxf(cm, __shfl_xor(cm, msk, 32));
            const float mnew = fmaxf(mrun[r], cm);
            const float sc = __expf(mrun[r] - mnew);
            const float p0 = __expf(s[0][r] - mnew);
            const float p1 = __expf(s[1][r] - mnew);
            float rs = p0 + p1;
#pragma unroll
            for (int msk = 1; msk < 16; msk <<= 1)
                rs += __shfl_xor(rs, msk, 32);
            lrun[r] = lrun[r] * sc + rs;
            mrun[r] = mnew;
            o[0][r] *= sc; o[1][r] *= sc; o[2][r] *= sc; o[3][r] *= sc;
            myp[(r + 8 * half) * 32 + n]      = (_Float16)p0;
            myp[(r + 8 * half) * 32 + 16 + n] = (_Float16)p1;
        }
        // wave-private LDS: wait for DS stores before re-reading as A-frag
        asm volatile("s_wait_dscnt 0" ::: "memory");
        v16h ap = load_frag_a(myp, 32, lane);
        // ---- O += P * V (vt rows contiguous over keys) -------------------
#pragma unroll
        for (int j = 0; j < 4; ++j) {
            const _Float16* vb =
                vt + ((long)((b * Hn + h) * Dh + j * 16)) * Tdim + kk;
            v16h bv = load_frag_b(vb, Tdim, lane);
            o[j] = wmma_f16(ap, bv, o[j]);
        }
    }

#pragma unroll
    for (int j = 0; j < 4; ++j) {
#pragma unroll
        for (int r = 0; r < 8; ++r) {
            const float val = o[j][r] / lrun[r];
            y[((long)(b * Tdim + q0 + r + 8 * half)) * Cdim + h * Dh + j * 16 + n] =
                (_Float16)val;
        }
    }
}

// ---------------------------------------------------------------------------
// Host-side orchestration
// ---------------------------------------------------------------------------
extern "C" void kernel_launch(void* const* d_in, const int* in_sizes, int n_in,
                              void* d_out, int out_size, void* d_ws, size_t ws_size,
                              hipStream_t stream) {
    const float* x        = (const float*)d_in[0];
    const float* w_attn   = (const float*)d_in[1];
    const float* b_attn   = (const float*)d_in[2];
    const float* w_proj   = (const float*)d_in[3];
    const float* b_proj   = (const float*)d_in[4];
    const float* w_fc     = (const float*)d_in[5];
    const float* b_fc     = (const float*)d_in[6];
    const float* w_mlpp   = (const float*)d_in[7];
    const float* b_mlpp   = (const float*)d_in[8];
    float* out = (float*)d_out;

    char* ws = (char*)d_ws;
    size_t off = 0;
    auto alloc = [&](size_t bytes) -> char* {
        char* p = ws + off;
        off += (bytes + 255) & ~(size_t)255;
        return p;
    };

    _Float16* wattn_h = (_Float16*)alloc((size_t)ThreeC * Cdim * 2);
    _Float16* wproj_h = (_Float16*)alloc((size_t)Cdim * Cdim * 2);
    _Float16* wfc_h   = (_Float16*)alloc((size_t)FourC * Cdim * 2);
    _Float16* wmlp_h  = (_Float16*)alloc((size_t)Cdim * FourC * 2);
    _Float16* xhat    = (_Float16*)alloc((size_t)Mrows * Cdim * 2);
    _Float16* qkv     = (_Float16*)alloc((size_t)Mrows * ThreeC * 2);
    _Float16* vtb     = (_Float16*)alloc((size_t)Bdim * Hn * Dh * Tdim * 2);
    _Float16* ybuf    = (_Float16*)alloc((size_t)Mrows * Cdim * 2);
    float*    x1      = (float*)   alloc((size_t)Mrows * Cdim * 4);
    _Float16* x1hat   = (_Float16*)alloc((size_t)Mrows * Cdim * 2);
    _Float16* hbuf    = (_Float16*)alloc((size_t)Mrows * FourC * 2);

    auto cast = [&](const float* src, _Float16* dst, long n) {
        int blocks = (int)((n + 255) / 256);
        cast_f32_to_f16_kernel<<<blocks, 256, 0, stream>>>(src, dst, n);
    };
    cast(w_attn, wattn_h, (long)ThreeC * Cdim);
    cast(w_proj, wproj_h, (long)Cdim * Cdim);
    cast(w_fc,   wfc_h,   (long)FourC * Cdim);
    cast(w_mlpp, wmlp_h,  (long)Cdim * FourC);

    // ln1(x) -> xhat
    ln_f16_kernel<<<(int)Mrows, 256, 0, stream>>>(x, xhat);

    // qkv = xhat @ w_attn^T + b_attn  [8192 x 3072]
    wmma_gemm_kernel<0><<<dim3(ThreeC / 256, (int)(Mrows / 128)), 256, 0, stream>>>(
        xhat, Cdim, wattn_h, Cdim, b_attn, nullptr, qkv, ThreeC, Cdim);

    // V transpose for P*V B-fragments
    {
        long n = (long)Bdim * Hn * Dh * Tdim;
        vtrans_kernel<<<(int)(n / 256), 256, 0, stream>>>(qkv, vtb);
    }

    // attention -> ybuf f16
    attn_kernel<<<dim3(Tdim / 128, Hn, Bdim), 256, 0, stream>>>(qkv, vtb, ybuf);

    // x1 = ybuf @ w_proj^T + b_proj + x   (f32)
    wmma_gemm_kernel<2><<<dim3(Cdim / 256, (int)(Mrows / 128)), 256, 0, stream>>>(
        ybuf, Cdim, wproj_h, Cdim, b_proj, x, x1, Cdim, Cdim);

    // ln2(x1) -> x1hat
    ln_f16_kernel<<<(int)Mrows, 256, 0, stream>>>(x1, x1hat);

    // hbuf = gelu(x1hat @ w_fc^T + b_fc)  [8192 x 4096]
    wmma_gemm_kernel<1><<<dim3(FourC / 256, (int)(Mrows / 128)), 256, 0, stream>>>(
        x1hat, Cdim, wfc_h, Cdim, b_fc, nullptr, hbuf, FourC, Cdim);

    // out = hbuf @ w_mlp_proj^T + b_mlp_proj + x1  (f32)
    wmma_gemm_kernel<2><<<dim3(Cdim / 256, (int)(Mrows / 128)), 256, 0, stream>>>(
        hbuf, FourC, wmlp_h, FourC, b_mlpp, x1, out, Cdim, FourC);

    (void)in_sizes; (void)n_in; (void)out_size; (void)ws_size;
}
